// SlidingWindowAttention_71279277245026
// MI455X (gfx1250) — matrix operand order repaired and checked by reading, not compiled, not run on hardware
//
#include <hip/hip_runtime.h>
#include <hip/hip_bf16.h>

// ---------------- problem constants ----------------
#define HIDDEN 2048
#define S_LEN  4096
#define NH     16
#define NKV    4
#define HD     128
#define WINDOW 2048

typedef __bf16 bf16;
typedef bf16  v16bf __attribute__((ext_vector_type(16)));
typedef bf16  v8bf  __attribute__((ext_vector_type(8)));
typedef float v8f   __attribute__((ext_vector_type(8)));

static __device__ __forceinline__ unsigned short f2bf(float f) {
    unsigned u = __builtin_bit_cast(unsigned, f);
    u = u + 0x7FFFu + ((u >> 16) & 1u);   // round-to-nearest-even
    return (unsigned short)(u >> 16);
}

static __device__ __forceinline__ v16bf cat8(v8bf lo, v8bf hi) {
    return __builtin_shufflevector(lo, hi, 0,1,2,3,4,5,6,7,8,9,10,11,12,13,14,15);
}

#define WMMA_BF16(a, b, c) \
    __builtin_amdgcn_wmma_f32_16x16x32_bf16(false, (a), false, (b), (short)0, (c), false, false)

// ---------------- f32 -> bf16 convert ----------------
__global__ void swa_cvt_bf16(const float* __restrict__ in,
                             unsigned short* __restrict__ out, int n) {
    int i = blockIdx.x * blockDim.x + threadIdx.x;
    if (i < n) out[i] = f2bf(in[i]);
}

// ---------------- bf16 GEMM: C[M,N] = A[M,K] * B[N,K]^T ----------------
// A, B row-major, K contiguous. One 16x64 output strip per wave:
// one A operand reused against 4 B operands (4 accumulators).
__global__ void __launch_bounds__(256)
swa_gemm_bf16(const unsigned short* __restrict__ A,
              const unsigned short* __restrict__ B,
              float* __restrict__ C, int M, int N, int K) {
    const int wave = threadIdx.x >> 5;
    const int lane = threadIdx.x & 31;
    const int ln = lane & 15;
    const int hi = lane >> 4;

    const int strips_n = N >> 6;                       // groups of 4 n-tiles
    const int unit = blockIdx.x * 8 + wave;
    if (unit >= (M >> 4) * strips_n) return;           // wave-uniform exit
    const int tm = unit / strips_n;
    const int tn4 = unit - tm * strips_n;

    const unsigned short* arow = A + (size_t)(tm * 16 + ln) * K;
    const unsigned short* brow = B + (size_t)(tn4 * 64 + ln) * K;
    const size_t bstride = (size_t)16 * K;             // 16 rows of B per n-tile

    v8f acc[4];
#pragma unroll
    for (int j = 0; j < 4; ++j) acc[j] = v8f{};

    for (int k0 = 0; k0 < K; k0 += 32) {
        v16bf av = *(const v16bf*)(arow + k0 + hi * 16);
#pragma unroll
        for (int j = 0; j < 4; ++j) {
            v16bf bv = *(const v16bf*)(brow + j * bstride + k0 + hi * 16);
            acc[j] = WMMA_BF16(av, bv, acc[j]);
        }
    }
    float* crow = C + (size_t)(tm * 16 + hi * 8) * N + tn4 * 64 + ln;
#pragma unroll
    for (int j = 0; j < 4; ++j)
#pragma unroll
        for (int r = 0; r < 8; ++r)
            crow[(size_t)r * N + j * 16] = acc[j][r];
}

// ---------------- RoPE on Q (also folds in softmax scale 1/sqrt(HD)) ------------
// Qraw[S,2048] f32 -> Qbf[NH][S][HD] bf16
__global__ void swa_rope_q(const float* __restrict__ Qraw,
                           unsigned short* __restrict__ Qbf) {
    int i = blockIdx.x * blockDim.x + threadIdx.x;          // NH*S*64
    if (i >= NH * S_LEN * 64) return;
    int d = i & 63;
    int s = (i >> 6) & (S_LEN - 1);
    int h = i >> 18;
    // inv_freq = 10000^(-2d/128); ln(10000)=9.210340371976184
    float inv = expf(-(float)(2 * d) * (9.210340371976184f / (float)HD));
    float ang = (float)s * inv;
    float c = cosf(ang), sn = sinf(ang);
    const float scale = 0.08838834764831845f;               // 1/sqrt(128)
    const float* q = Qraw + (size_t)s * HIDDEN + h * HD;
    float x1 = q[d], x2 = q[d + 64];
    unsigned short* o = Qbf + ((size_t)h * S_LEN + s) * HD;
    o[d]      = f2bf((x1 * c - x2 * sn) * scale);
    o[d + 64] = f2bf((x2 * c + x1 * sn) * scale);
}

// ---------------- RoPE on K: Kraw[S,512] f32 -> Kbf[NKV][S][HD] bf16 ----------------
__global__ void swa_rope_k(const float* __restrict__ Kraw,
                           unsigned short* __restrict__ Kbf) {
    int i = blockIdx.x * blockDim.x + threadIdx.x;          // NKV*S*64
    if (i >= NKV * S_LEN * 64) return;
    int d = i & 63;
    int s = (i >> 6) & (S_LEN - 1);
    int h = i >> 18;
    float inv = expf(-(float)(2 * d) * (9.210340371976184f / (float)HD));
    float ang = (float)s * inv;
    float c = cosf(ang), sn = sinf(ang);
    const float* k = Kraw + (size_t)s * (NKV * HD) + h * HD;
    float x1 = k[d], x2 = k[d + 64];
    unsigned short* o = Kbf + ((size_t)h * S_LEN + s) * HD;
    o[d]      = f2bf(x1 * c - x2 * sn);
    o[d + 64] = f2bf(x2 * c + x1 * sn);
}

// ---------------- V transpose: Vraw[S,512] f32 -> Vt[NKV][HD][S] bf16 ----------------
__global__ void swa_pack_vt(const float* __restrict__ Vraw,
                            unsigned short* __restrict__ Vt) {
    int i = blockIdx.x * blockDim.x + threadIdx.x;          // NKV*S*HD
    if (i >= NKV * S_LEN * HD) return;
    int d = i & 127;
    int s = (i >> 7) & (S_LEN - 1);
    int h = i >> 19;
    Vt[((size_t)h * HD + d) * S_LEN + s] =
        f2bf(Vraw[(size_t)s * (NKV * HD) + h * HD + d]);
}

// ---------------- flash sliding-window attention (one 16-row q tile per wave) ----------
__global__ void __launch_bounds__(256)
swa_attn(const unsigned short* __restrict__ Qbf,
         const unsigned short* __restrict__ Kbf,
         const unsigned short* __restrict__ Vt,
         unsigned short* __restrict__ Abf) {
    __shared__ __align__(32) unsigned short smem[8][16 * 32];  // per-wave P tile

    const int wave = threadIdx.x >> 5;
    const int lane = threadIdx.x & 31;
    const int ln = lane & 15;
    const int hi = lane >> 4;

    const int unit = blockIdx.x * 8 + wave;  // 4096 units = 256 qtiles * 16 heads
    const int head = unit & 15;
    const int qt = unit >> 4;
    const int kvh = head >> 2;
    const int q0 = qt * 16;

    // Q tile as 4 A-operands (d chunks of 32); scale already folded in
    const unsigned short* Qh = Qbf + ((size_t)head * S_LEN + q0 + ln) * HD;
    v16bf qa[4];
#pragma unroll
    for (int c = 0; c < 4; ++c) {
        qa[c] = *(const v16bf*)(Qh + c * 32 + hi * 16);
    }

    v8f O[8];
#pragma unroll
    for (int t = 0; t < 8; ++t) O[t] = v8f{};
    float Mr[8], Lr[8];
#pragma unroll
    for (int r = 0; r < 8; ++r) { Mr[r] = -1e30f; Lr[r] = 0.0f; }

    int kstart = q0 - (WINDOW - 1); if (kstart < 0) kstart = 0;
    const int kb0 = kstart & ~31;
    const int kend = q0 + 15;

    const unsigned short* Kh = Kbf + (size_t)kvh * S_LEN * HD;
    const unsigned short* Vh = Vt + (size_t)kvh * HD * S_LEN;
    unsigned short* sm = smem[wave];

    const float L2E = 1.4426950408889634f;

    for (int kb = kb0; kb <= kend; kb += 32) {
        // scores for 32 keys: two 16x16 tiles
        v8f s0 = {}, s1 = {};
#pragma unroll
        for (int c = 0; c < 4; ++c) {
            v16bf b0 = *(const v16bf*)(Kh + (size_t)(kb + ln) * HD + c * 32 + hi * 16);
            v16bf b1 = *(const v16bf*)(Kh + (size_t)(kb + 16 + ln) * HD + c * 32 + hi * 16);
            s0 = WMMA_BF16(qa[c], b0, s0);
            s1 = WMMA_BF16(qa[c], b1, s1);
        }

        // online softmax per row (rows r+hi*8 live in this lane group)
        float p0[8], p1[8];
#pragma unroll
        for (int r = 0; r < 8; ++r) {
            const int qrow = q0 + r + hi * 8;
            const int k0i = kb + ln;
            const int k1i = kb + 16 + ln;
            const bool a0 = (k0i <= qrow) && (qrow - k0i < WINDOW);
            const bool a1 = (k1i <= qrow) && (qrow - k1i < WINDOW);
            float v0 = a0 ? s0[r] : -1e30f;
            float v1 = a1 ? s1[r] : -1e30f;
            float mx = fmaxf(v0, v1);
#pragma unroll
            for (int off = 1; off < 16; off <<= 1)
                mx = fmaxf(mx, __shfl_xor(mx, off, 32));
            const float newM = fmaxf(Mr[r], mx);
            const float alpha = exp2f((Mr[r] - newM) * L2E);
            float e0 = a0 ? exp2f((v0 - newM) * L2E) : 0.0f;
            float e1 = a1 ? exp2f((v1 - newM) * L2E) : 0.0f;
            float rs = e0 + e1;
#pragma unroll
            for (int off = 1; off < 16; off <<= 1)
                rs += __shfl_xor(rs, off, 32);
            Lr[r] = Lr[r] * alpha + rs;
            Mr[r] = newM;
            p0[r] = e0; p1[r] = e1;
#pragma unroll
            for (int dt = 0; dt < 8; ++dt) O[dt][r] *= alpha;
        }

        // transpose P (C-layout) -> A-layout through per-wave LDS
#pragma unroll
        for (int r = 0; r < 8; ++r) {
            sm[(r + hi * 8) * 32 + ln]      = f2bf(p0[r]);
            sm[(r + hi * 8) * 32 + 16 + ln] = f2bf(p1[r]);
        }
        asm volatile("s_wait_dscnt 0" ::: "memory");
        v16bf pa = *(const v16bf*)(sm + ln * 32 + hi * 16);

        // O += P * V  (Vt rows are K-contiguous over keys)
#pragma unroll
        for (int dt = 0; dt < 8; ++dt) {
            v16bf vb = *(const v16bf*)(Vh + (size_t)(dt * 16 + ln) * S_LEN + kb + hi * 16);
            O[dt] = WMMA_BF16(pa, vb, O[dt]);
        }
    }

    // normalize + store bf16 to [S, NH*HD]
    unsigned short* outp = Abf + (size_t)(q0 + hi * 8) * HIDDEN + head * HD + ln;
#pragma unroll
    for (int r = 0; r < 8; ++r) {
        const float rl = 1.0f / Lr[r];
#pragma unroll
        for (int dt = 0; dt < 8; ++dt)
            outp[(size_t)r * HIDDEN + dt * 16] = f2bf(O[dt][r] * rl);
    }
}

// ---------------- host launch ----------------
extern "C" void kernel_launch(void* const* d_in, const int* in_sizes, int n_in,
                              void* d_out, int out_size, void* d_ws, size_t ws_size,
                              hipStream_t stream) {
    const float* hidden = (const float*)d_in[0];   // [1, 4096, 2048]
    const float* q_w    = (const float*)d_in[1];   // [2048, 2048]
    const float* k_w    = (const float*)d_in[2];   // [512, 2048]
    const float* v_w    = (const float*)d_in[3];   // [512, 2048]
    const float* o_w    = (const float*)d_in[4];   // [2048, 2048]
    float* out = (float*)d_out;                    // [4096, 2048] f32

    // ---- workspace carve ----
    char* p = (char*)d_ws;
    auto carve = [&](size_t bytes) -> void* {
        void* r = (void*)p;
        p += (bytes + 255) & ~(size_t)255;
        return r;
    };
    const size_t nX  = (size_t)S_LEN * HIDDEN;          // 8M
    const size_t nQW = (size_t)HIDDEN * HIDDEN;         // 4M
    const size_t nKW = (size_t)(NKV * HD) * HIDDEN;     // 1M

    unsigned short* Xbf  = (unsigned short*)carve(nX * 2);
    unsigned short* Wq   = (unsigned short*)carve(nQW * 2);
    unsigned short* Wk   = (unsigned short*)carve(nKW * 2);
    unsigned short* Wv   = (unsigned short*)carve(nKW * 2);
    unsigned short* Wo   = (unsigned short*)carve(nQW * 2);
    float*          Qraw = (float*)carve(nX * 4);
    float*          Kraw = (float*)carve((size_t)S_LEN * NKV * HD * 4);
    float*          Vraw = (float*)carve((size_t)S_LEN * NKV * HD * 4);
    unsigned short* Qbf  = (unsigned short*)carve((size_t)NH  * S_LEN * HD * 2);
    unsigned short* Kbf  = (unsigned short*)carve((size_t)NKV * S_LEN * HD * 2);
    unsigned short* Vt   = (unsigned short*)carve((size_t)NKV * HD * S_LEN * 2);
    unsigned short* Abf  = (unsigned short*)carve(nX * 2);

    const int CT = 256;
    auto cvb = [&](size_t n) { return (int)((n + CT - 1) / CT); };

    // 1) convert inputs to bf16
    swa_cvt_bf16<<<cvb(nX),  CT, 0, stream>>>(hidden, Xbf, (int)nX);
    swa_cvt_bf16<<<cvb(nQW), CT, 0, stream>>>(q_w, Wq, (int)nQW);
    swa_cvt_bf16<<<cvb(nKW), CT, 0, stream>>>(k_w, Wk, (int)nKW);
    swa_cvt_bf16<<<cvb(nKW), CT, 0, stream>>>(v_w, Wv, (int)nKW);
    swa_cvt_bf16<<<cvb(nQW), CT, 0, stream>>>(o_w, Wo, (int)nQW);

    // 2) QKV projections (WMMA GEMMs), 16x64 strip per wave
    {
        int blocks = (S_LEN / 16) * (HIDDEN / 64) / 8;       // 1024
        swa_gemm_bf16<<<blocks, 256, 0, stream>>>(Xbf, Wq, Qraw, S_LEN, HIDDEN, HIDDEN);
    }
    {
        int blocks = (S_LEN / 16) * ((NKV * HD) / 64) / 8;   // 256
        swa_gemm_bf16<<<blocks, 256, 0, stream>>>(Xbf, Wk, Kraw, S_LEN, NKV * HD, HIDDEN);
        swa_gemm_bf16<<<blocks, 256, 0, stream>>>(Xbf, Wv, Vraw, S_LEN, NKV * HD, HIDDEN);
    }

    // 3) RoPE + repack
    swa_rope_q <<<cvb((size_t)NH  * S_LEN * 64), CT, 0, stream>>>(Qraw, Qbf);
    swa_rope_k <<<cvb((size_t)NKV * S_LEN * 64), CT, 0, stream>>>(Kraw, Kbf);
    swa_pack_vt<<<cvb((size_t)NKV * S_LEN * HD), CT, 0, stream>>>(Vraw, Vt);

    // 4) flash sliding-window attention
    swa_attn<<<(256 * NH) / 8, 256, 0, stream>>>(Qbf, Kbf, Vt, Abf);   // 512 blocks

    // 5) output projection -> f32 result
    {
        int blocks = (S_LEN / 16) * (HIDDEN / 64) / 8;
        swa_gemm_bf16<<<blocks, 256, 0, stream>>>(Abf, Wo, out, S_LEN, HIDDEN, HIDDEN);
    }
}
